// FullAttention_2052994367610
// MI455X (gfx1250) — matrix operand order repaired and checked by reading, not compiled, not run on hardware
//
#include <hip/hip_runtime.h>
#include <hip/hip_bf16.h>

typedef __attribute__((ext_vector_type(16))) _Float16 v16h;
typedef __attribute__((ext_vector_type(8)))  _Float16 v8h;
typedef __attribute__((ext_vector_type(8)))  float    v8f;

static constexpr int   Bc = 4, Lc = 2048, Hc = 8, Ec = 64;
static constexpr float SCALE = 0.125f;     // 1/sqrt(64)
static constexpr float NEGF  = -1e30f;

__device__ __forceinline__ v8f wmma_f16(v16h a, v16h b, v8f c) {
  // D = A(16x32 f16) * B(32x16 f16) + C(16x16 f32)
  return __builtin_amdgcn_wmma_f32_16x16x32_f16(
      /*neg_a=*/false, a, /*neg_b=*/false, b,
      /*c_mod=*/(short)0, c, /*reuse_a=*/false, /*reuse_b=*/false);
}

__device__ __forceinline__ v16h cat8(v8h lo, v8h hi) {
  return __builtin_shufflevector(lo, hi, 0,1,2,3,4,5,6,7,8,9,10,11,12,13,14,15);
}

// 16 contiguous halves (32B) from LDS as a WMMA operand (2x ds_load_b128)
__device__ __forceinline__ v16h lds_load16(const _Float16* p) {
  v8h lo = *(const v8h*)p;
  v8h hi = *(const v8h*)(p + 8);
  return cat8(lo, hi);
}

// Reductions across the 16 lanes of one half-wave (xor masks < 16 stay in-half)
__device__ __forceinline__ float rowmax16(float v) {
  v = fmaxf(v, __shfl_xor(v, 1, 32));
  v = fmaxf(v, __shfl_xor(v, 2, 32));
  v = fmaxf(v, __shfl_xor(v, 4, 32));
  v = fmaxf(v, __shfl_xor(v, 8, 32));
  return v;
}
__device__ __forceinline__ float rowsum16(float v) {
  v += __shfl_xor(v, 1, 32);
  v += __shfl_xor(v, 2, 32);
  v += __shfl_xor(v, 4, 32);
  v += __shfl_xor(v, 8, 32);
  return v;
}

__global__ __launch_bounds__(256)
void fa_fwd_causal_kernel(const float* __restrict__ Q,
                          const float* __restrict__ K,
                          const float* __restrict__ V,
                          float* __restrict__ O) {
  // Double-buffered fp32 staging filled by the async DMA path, plus f16 operand tiles.
  __shared__ __align__(16) float    sKf[2][32 * 64];   // [buf][key][dim]
  __shared__ __align__(16) float    sVf[2][32 * 64];   // [buf][key][dim]
  __shared__ __align__(16) _Float16 sK16[32 * 64];     // [key][dim]   row-major
  __shared__ __align__(16) _Float16 sVT16[64 * 32];    // [dim][key]   transposed
  __shared__ __align__(16) _Float16 sP[8][16 * 32];    // per-wave P tile

  const int tid    = threadIdx.x;
  const int wave   = tid >> 5;
  const int lane   = tid & 31;
  const int halfid = lane >> 4;   // 0: lanes 0-15, 1: lanes 16-31
  const int ln     = lane & 15;

  const int qbase = blockIdx.x * 128;
  const int h     = blockIdx.y;
  const int b     = blockIdx.z;
  const int q0    = qbase + wave * 16;

  // Issue one chunk (32 keys x 64 dims, fp32) of K and V into LDS via async DMA.
  // 512 float4 per tensor / 256 threads = 2 each -> exactly 4 async ops per wave.
  auto issue_copy = [&](int c, int buf) {
#pragma unroll
    for (int j = 0; j < 2; ++j) {
      const int i4  = tid + j * 256;          // float4 index 0..511
      const int key = i4 >> 4;                // 16 float4 per key row
      const int f4  = i4 & 15;
      const size_t g = (((size_t)b * Lc + (c + key)) * Hc + h) * Ec + f4 * 4;
      const unsigned lk = (unsigned)(uintptr_t)&sKf[buf][key * 64 + f4 * 4];
      const unsigned lv = (unsigned)(uintptr_t)&sVf[buf][key * 64 + f4 * 4];
      asm volatile("global_load_async_to_lds_b128 %0, %1, off"
                   :: "v"(lk), "v"(K + g) : "memory");
      asm volatile("global_load_async_to_lds_b128 %0, %1, off"
                   :: "v"(lv), "v"(V + g) : "memory");
    }
  };

  // ---- Q tile -> WMMA A-layout registers (16x32 f16 per 32-dim chunk of E=64)
  const float* qp = Q + (((size_t)b * Lc + (q0 + ln)) * Hc + h) * Ec;
  v16h aq0, aq1;
  {
    const int d0 = halfid * 16;
#pragma unroll
    for (int i = 0; i < 8; ++i) {
      aq0[i]     = (_Float16)qp[d0 + i];
      aq0[i + 8] = (_Float16)qp[8 + d0 + i];
      aq1[i]     = (_Float16)qp[32 + d0 + i];
      aq1[i + 8] = (_Float16)qp[40 + d0 + i];
    }
  }

  v8f acc0 = {}, acc1 = {}, acc2 = {}, acc3 = {};
  float m[8], l[8];
#pragma unroll
  for (int r = 0; r < 8; ++r) { m[r] = NEGF; l[r] = 0.f; }

  const int cend = qbase + 128;
  issue_copy(0, 0);                 // preload chunk 0 into buffer 0
  int p = 0;

  for (int c = 0; c < cend; c += 32) {
    const bool have_next = (c + 32 < cend);     // block-uniform
    if (have_next) {
      issue_copy(c + 32, p ^ 1);                // overlap DMA with this chunk's compute
      if (c + 64 < cend) {                      // warm L2 two chunks ahead
        __builtin_prefetch(&K[(((size_t)b * Lc + (c + 64 + lane)) * Hc + h) * Ec], 0, 1);
        __builtin_prefetch(&V[(((size_t)b * Lc + (c + 64 + lane)) * Hc + h) * Ec], 0, 1);
      }
      // 8 async ops outstanding; in-order retire => <=4 means chunk c has landed
      asm volatile("s_wait_asynccnt 0x4" ::: "memory");
    } else {
      asm volatile("s_wait_asynccnt 0x0" ::: "memory");
    }
    __syncthreads();

    // ---- cooperative convert: fp32 staging -> f16 operand tiles (V transposed)
#pragma unroll
    for (int j = 0; j < 8; ++j) {
      const int idx = tid + j * 256;
      const int key = idx >> 6, dim = idx & 63;
      const float kv = sKf[p][key * 64 + dim];
      const float vv = sVf[p][key * 64 + dim];
      sK16[key * 64 + dim]  = (_Float16)kv;
      sVT16[dim * 32 + key] = (_Float16)vv;
    }
    __syncthreads();

    if (c <= q0 + 15) {   // wave-uniform: EXEC all-ones around WMMA
      // ---- S = Q * K^T (two 16-key tiles, K split 32+32)
      v8f s0 = {}, s1 = {};
      s0 = wmma_f16(aq0, lds_load16(&sK16[(0 * 16 + ln) * 64 +  0 + halfid * 16]), s0);
      s0 = wmma_f16(aq1, lds_load16(&sK16[(0 * 16 + ln) * 64 + 32 + halfid * 16]), s0);
      s1 = wmma_f16(aq0, lds_load16(&sK16[(1 * 16 + ln) * 64 +  0 + halfid * 16]), s1);
      s1 = wmma_f16(aq1, lds_load16(&sK16[(1 * 16 + ln) * 64 + 32 + halfid * 16]), s1);

      // ---- online softmax (C layout: row = r + 8*halfid, col = ln / 16+ln)
      float p0[8], p1[8], alpha[8];
#pragma unroll
      for (int r = 0; r < 8; ++r) {
        const int qr = q0 + r + 8 * halfid;
        const float t0 = ((c + ln)      <= qr) ? s0[r] * SCALE : NEGF;
        const float t1 = ((c + 16 + ln) <= qr) ? s1[r] * SCALE : NEGF;
        const float tm = rowmax16(fmaxf(t0, t1));
        const float mn = fmaxf(m[r], tm);
        alpha[r] = __expf(m[r] - mn);
        p0[r]    = __expf(t0 - mn);
        p1[r]    = __expf(t1 - mn);
        l[r] = l[r] * alpha[r] + rowsum16(p0[r] + p1[r]);
        m[r] = mn;
      }
#pragma unroll
      for (int r = 0; r < 8; ++r) {
        acc0[r] *= alpha[r]; acc1[r] *= alpha[r];
        acc2[r] *= alpha[r]; acc3[r] *= alpha[r];
      }

      // ---- P: C-layout -> A-layout via per-wave LDS round trip (DS is in-order)
#pragma unroll
      for (int r = 0; r < 8; ++r) {
        const int row = r + 8 * halfid;
        sP[wave][row * 32 + ln]      = (_Float16)p0[r];
        sP[wave][row * 32 + 16 + ln] = (_Float16)p1[r];
      }
      const v16h ap = cat8(*(const v8h*)&sP[wave][ln * 32 + halfid * 16],
                           *(const v8h*)&sP[wave][ln * 32 + halfid * 16 + 8]);

      // ---- O += P(16x32) * V(32x64)
      acc0 = wmma_f16(ap, lds_load16(&sVT16[(0 * 16 + ln) * 32 + halfid * 16]), acc0);
      acc1 = wmma_f16(ap, lds_load16(&sVT16[(1 * 16 + ln) * 32 + halfid * 16]), acc1);
      acc2 = wmma_f16(ap, lds_load16(&sVT16[(2 * 16 + ln) * 32 + halfid * 16]), acc2);
      acc3 = wmma_f16(ap, lds_load16(&sVT16[(3 * 16 + ln) * 32 + halfid * 16]), acc3);
    }
    __syncthreads();
    p ^= 1;
  }

  // ---- epilogue: O[b, qr, h, :] = acc / l
#pragma unroll
  for (int r = 0; r < 8; ++r) {
    const int qr = q0 + r + 8 * halfid;
    const float inv = 1.0f / l[r];
    float* op = O + (((size_t)b * Lc + qr) * Hc + h) * Ec + ln;
    op[0]  = acc0[r] * inv;
    op[16] = acc1[r] * inv;
    op[32] = acc2[r] * inv;
    op[48] = acc3[r] * inv;
  }
}

extern "C" void kernel_launch(void* const* d_in, const int* in_sizes, int n_in,
                              void* d_out, int out_size, void* d_ws, size_t ws_size,
                              hipStream_t stream) {
  (void)in_sizes; (void)n_in; (void)out_size; (void)d_ws; (void)ws_size;
  const float* Q = (const float*)d_in[0];
  const float* K = (const float*)d_in[1];
  const float* V = (const float*)d_in[2];
  float* O = (float*)d_out;
  dim3 grid(Lc / 128, Hc, Bc);   // (16, 8, 4): q-block, head, batch
  fa_fwd_causal_kernel<<<grid, 256, 0, stream>>>(Q, K, V, O);
}
